// Generator_39213051412777
// MI455X (gfx1250) — matrix operand order repaired and checked by reading, not compiled, not run on hardware
//
#include <hip/hip_runtime.h>
#include <hip/hip_bf16.h>

// ---------------------------------------------------------------------------
// CDNA5 / gfx1250 bf16 WMMA GEMM pipeline for the GRU/highway generator.
// Both A (activations) and B (weights) are packed into one shared
// v_wmma_f32_16x16x32_bf16 per-lane fragment order (k = 16*(lane/16) + j), so the GEMM hot loop is
// pure b128 loads + prefetch + WMMA (no conversions, no layout shuffles).
// ---------------------------------------------------------------------------

typedef __attribute__((ext_vector_type(16))) __bf16 v16bf;
typedef __attribute__((ext_vector_type(8)))  float  v8f;

__device__ __forceinline__ __bf16 f2bf(float f) {
  unsigned u = __builtin_bit_cast(unsigned, f);
  unsigned r = u + 0x7FFFu + ((u >> 16) & 1u);     // round-to-nearest-even
  unsigned short h = (unsigned short)(r >> 16);
  return __builtin_bit_cast(__bf16, h);
}

// ---------------------------------------------------------------------------
// B packer: fragments for 16x16x32 bf16 WMMA, B = 32(K) x 16(N) tiles.
// Packed idx = ((kt*(Ntot/16) + nt)*32 + lane)*16 + j with
//   lane 0..15  -> N = nt*16 + lane,      K = kt*32 + j
//   lane 16..31 -> N = nt*16 + lane-16,   K = kt*32 + 16 + j
// Source W is (Nsrc x K) row-major if transpose (we need W^T), else (K x Nsrc).
// nOff places the source at column offset inside a wider packed matrix (Ntot).
// ---------------------------------------------------------------------------
__global__ void pack_b_kernel(const float* __restrict__ W, __bf16* __restrict__ out,
                              int K, int Nsrc, int Ntot, int nOff, int transpose) {
  long i = (long)blockIdx.x * blockDim.x + threadIdx.x;
  long total = (long)K * Nsrc;
  if (i >= total) return;
  int  ns  = (int)(i % Nsrc);
  long k   = i / Nsrc;
  int  n   = ns + nOff;
  long kt  = k >> 5;
  int  klo = (int)(k & 31);
  int  lane = ((klo >> 4) << 4) + (n & 15);
  int  j    = klo & 15;
  long idx  = ((kt * (Ntot >> 4) + (n >> 4)) * 32 + lane) * 16 + j;
  float v = transpose ? W[(long)ns * K + k] : W[k * (long)Nsrc + ns];
  out[idx] = f2bf(v);
}

// ---------------------------------------------------------------------------
// A packer: fragments for 16x16x32 bf16 WMMA, A = 16(M) x 32(K) tiles.
// Packed idx = ((kt*(M/16) + mt)*32 + lane)*16 + j with
//   lane 0..15  -> M = mt*16 + lane,    K = kt*32 + j
//   lane 16..31 -> M = mt*16 + lane-16, K = kt*32 + 16 + j
// ---------------------------------------------------------------------------
__global__ void pack_a_kernel(const float* __restrict__ A, __bf16* __restrict__ out,
                              int M, int K) {
  long i = (long)blockIdx.x * blockDim.x + threadIdx.x;
  long total = (long)M * K;
  if (i >= total) return;
  int  j    = (int)(i & 15);
  int  lane = (int)((i >> 4) & 31);
  long tile = i >> 9;
  int  MT   = M >> 4;
  int  mt   = (int)(tile % MT);
  long kt   = tile / MT;
  int  m    = mt * 16 + (lane & 15);
  long k    = kt * 32 + ((lane >> 4) << 4) + j;
  out[i] = f2bf(A[(long)m * K + k]);
}

// ---------------------------------------------------------------------------
// GEMM: C[M x N] = Apacked(bf16) * Bpacked(bf16) (+ bias over columns)
// Block = 8 waves; wave computes 16(M) x 32(N); block tile = 32 x 128.
// grid = (N/128, M/32, KSLICES). atomic!=0 -> split-K atomicAdd into C.
// Inner loop: 3 contiguous 32B lane loads + global_prefetch + 2 WMMA.
// ---------------------------------------------------------------------------
__global__ __launch_bounds__(256) void gemm_bf16_kernel(
    const __bf16* __restrict__ Ap,
    const __bf16* __restrict__ Bp,
    const float* __restrict__ bias,
    float* __restrict__ C, long ldc,
    int M, int N, int K, int atomic) {
  const int w    = threadIdx.x >> 5;
  const int lane = threadIdx.x & 31;
  const int m0   = blockIdx.y * 32 + (w >> 2) * 16;
  const int n0   = blockIdx.x * 128 + (w & 3) * 32;
  const int NT   = N >> 4;
  const int MT   = M >> 4;
  const int mt   = m0 >> 4;

  // K-chunk range for this z-slice (split-K)
  const int chunks = K >> 5;
  const int per    = (chunks + (int)gridDim.z - 1) / (int)gridDim.z;
  int c0 = (int)blockIdx.z * per;
  int c1 = c0 + per; if (c1 > chunks) c1 = chunks;

  v8f acc0 = {}; v8f acc1 = {};
  const v16bf* apv = reinterpret_cast<const v16bf*>(Ap);
  const v16bf* bpv = reinterpret_cast<const v16bf*>(Bp);

#pragma unroll 2
  for (int c = c0; c < c1; ++c) {
    v16bf a  = apv[((long)c * MT + mt) * 32 + lane];
    long  t0 = (long)c * NT + (n0 >> 4);
    v16bf b0 = bpv[t0 * 32 + lane];
    v16bf b1 = bpv[(t0 + 1) * 32 + lane];
    // Speculative prefetch of next K-chunk's B tile (dropped if OOB).
    __builtin_prefetch(reinterpret_cast<const char*>(bpv + (t0 + NT) * 32 + lane), 0, 1);
    acc0 = __builtin_amdgcn_wmma_f32_16x16x32_bf16(false, a, false, b0,
                                                   (short)0, acc0, false, false);
    acc1 = __builtin_amdgcn_wmma_f32_16x16x32_bf16(false, a, false, b1,
                                                   (short)0, acc1, false, false);
  }

  // ---- store: lanes 0-15 -> M=m0+r, N=n0+lane; lanes 16-31 -> M=m0+8+r ----
  const int col  = n0 + (lane & 15);
  const int rowb = m0 + ((lane >> 4) << 3);
  const float bia0 = bias ? bias[col]      : 0.0f;
  const float bia1 = bias ? bias[col + 16] : 0.0f;
#pragma unroll
  for (int r = 0; r < 8; ++r) {
    const long i0 = (long)(rowb + r) * ldc + col;
    float v0 = acc0[r] + bia0;
    float v1 = acc1[r] + bia1;
    if (atomic) {
      atomicAdd(&C[i0], v0);
      atomicAdd(&C[i0 + 16], v1);
    } else {
      C[i0] = v0;
      C[i0 + 16] = v1;
    }
  }
}

// ---------------------------------------------------------------------------
// Elementwise kernels
// ---------------------------------------------------------------------------
__global__ void gru_update_kernel(const float* __restrict__ gi, const float* __restrict__ gh,
                                  const float* __restrict__ h_in, float* __restrict__ h_out,
                                  float* __restrict__ xcopy, int H) {
  const int b = blockIdx.y;
  const int h = blockIdx.x * blockDim.x + threadIdx.x;
  if (h >= H) return;
  const long g = (long)b * 3 * H;
  float ir = gi[g + h], iz = gi[g + H + h], in_ = gi[g + 2 * H + h];
  float hr = gh[g + h], hz = gh[g + H + h], hn  = gh[g + 2 * H + h];
  float r = 1.0f / (1.0f + __expf(-(ir + hr)));
  float u = 1.0f / (1.0f + __expf(-(iz + hz)));
  float n = tanhf(in_ + r * hn);
  float o = (1.0f - u) * n + u * h_in[(long)b * H + h];
  h_out[(long)b * H + h] = o;
  xcopy[(long)b * H + h] = o;
}

// cat holds [glin | hlin] per row (fused wg/wh GEMM output, row stride 2H)
__global__ void highway_update_kernel(const float* __restrict__ cat,
                                      float* __restrict__ x, int H) {
  const int b = blockIdx.y;
  const int h = blockIdx.x * blockDim.x + threadIdx.x;
  if (h >= H) return;
  const long cbase = (long)b * 2 * H;
  float g  = 1.0f / (1.0f + __expf(-cat[cbase + h]));
  float v  = cat[cbase + H + h];
  float nl = v > 0.0f ? v : (__expf(v) - 1.0f);   // elu
  const long i = (long)b * H + h;
  x[i] = g * nl + (1.0f - g) * x[i];
}

__global__ void softmax_gumbel_kernel(float* __restrict__ logits,
                                      const float* __restrict__ noise, int V) {
  const int b = blockIdx.x;
  __shared__ float red[256];
  float* row = logits + (long)b * V;
  const float* nr = noise + (long)b * V;
  float m = -3.0e38f;
  for (int i = threadIdx.x; i < V; i += 256) m = fmaxf(m, row[i] + nr[i]);
  red[threadIdx.x] = m; __syncthreads();
  for (int s = 128; s > 0; s >>= 1) {
    if (threadIdx.x < s) red[threadIdx.x] = fmaxf(red[threadIdx.x], red[threadIdx.x + s]);
    __syncthreads();
  }
  m = red[0]; __syncthreads();
  float sum = 0.0f;
  for (int i = threadIdx.x; i < V; i += 256) sum += __expf(row[i] + nr[i] - m);
  red[threadIdx.x] = sum; __syncthreads();
  for (int s = 128; s > 0; s >>= 1) {
    if (threadIdx.x < s) red[threadIdx.x] += red[threadIdx.x + s];
    __syncthreads();
  }
  const float inv = 1.0f / red[0];
  __syncthreads();
  for (int i = threadIdx.x; i < V; i += 256) row[i] = __expf(row[i] + nr[i] - m) * inv;
}

__global__ void concat_bias_kernel(const float* __restrict__ a, const float* __restrict__ b,
                                   float* __restrict__ out, int H) {
  const int i = blockIdx.x * blockDim.x + threadIdx.x;
  if (i < 2 * H) out[i] = (i < H) ? a[i] : b[i - H];
}

__global__ void fill_zero_kernel(float* __restrict__ p, long n) {
  long i = (long)blockIdx.x * blockDim.x + threadIdx.x;
  if (i < n) p[i] = 0.0f;
}

__global__ void scatter_out_kernel(const float* __restrict__ xe, float* __restrict__ out,
                                   int s, int S, int E) {
  const int b = blockIdx.y;
  const int e = blockIdx.x * blockDim.x + threadIdx.x;
  if (e < E) out[((long)b * S + s) * E + e] = xe[(long)b * E + e];
}

// ---------------------------------------------------------------------------
// Host driver
// ---------------------------------------------------------------------------
extern "C" void kernel_launch(void* const* d_in, const int* in_sizes, int n_in,
                              void* d_out, int out_size, void* d_ws, size_t ws_size,
                              hipStream_t stream) {
  (void)in_sizes; (void)n_in; (void)out_size;
  const int Bsz = 128, E = 512, H = 1024, V = 32000, S = 64;

  const float* x0       = (const float*)d_in[0];
  const float* z        = (const float*)d_in[1];
  const float* gumbel   = (const float*)d_in[2];
  const float* gru0_wih = (const float*)d_in[3];
  const float* gru0_whh = (const float*)d_in[4];
  const float* gru0_bih = (const float*)d_in[5];
  const float* gru0_bhh = (const float*)d_in[6];
  const float* gru1_wih = (const float*)d_in[7];
  const float* gru1_whh = (const float*)d_in[8];
  const float* gru1_bih = (const float*)d_in[9];
  const float* gru1_bhh = (const float*)d_in[10];
  const float* hw0_wg   = (const float*)d_in[11];
  const float* hw0_bg   = (const float*)d_in[12];
  const float* hw0_wh   = (const float*)d_in[13];
  const float* hw0_bh   = (const float*)d_in[14];
  const float* hw1_wg   = (const float*)d_in[15];
  const float* hw1_bg   = (const float*)d_in[16];
  const float* hw1_wh   = (const float*)d_in[17];
  const float* hw1_bh   = (const float*)d_in[18];
  const float* fc_w     = (const float*)d_in[19];
  const float* fc_b     = (const float*)d_in[20];
  const float* emb      = (const float*)d_in[21];
  float* out = (float*)d_out;

  // ---- workspace carve-out ----
  char* ws = (char*)d_ws;
  size_t off = 0;
  auto alloc = [&](size_t bytes) -> char* {
    char* p = ws + off;
    off = (off + bytes + 255) & ~(size_t)255;
    return p;
  };
  __bf16* pk_g0_wih  = (__bf16*)alloc((size_t)E * 3 * H * 2);
  __bf16* pk_g0_whh  = (__bf16*)alloc((size_t)H * 3 * H * 2);
  __bf16* pk_g1_wih  = (__bf16*)alloc((size_t)H * 3 * H * 2);
  __bf16* pk_g1_whh  = (__bf16*)alloc((size_t)H * 3 * H * 2);
  __bf16* pk_hw0_cat = (__bf16*)alloc((size_t)2 * H * 2 * H * 2);  // 2 layers, K=H, N=2H
  __bf16* pk_hw1_cat = (__bf16*)alloc((size_t)2 * H * 2 * H * 2);
  __bf16* pk_fc      = (__bf16*)alloc((size_t)H * V * 2);
  __bf16* pk_emb     = (__bf16*)alloc((size_t)V * E * 2);
  __bf16* apk        = (__bf16*)alloc((size_t)Bsz * V * 2);        // packed-A scratch (max K=V)
  float* hwb0   = (float*)alloc((size_t)2 * 2 * H * 4);            // concat biases, 2 layers
  float* hwb1   = (float*)alloc((size_t)2 * 2 * H * 4);
  float* gi     = (float*)alloc((size_t)Bsz * 3 * H * 4);
  float* gh     = (float*)alloc((size_t)Bsz * 3 * H * 4);
  float* h0a    = (float*)alloc((size_t)Bsz * H * 4);
  float* h0b    = (float*)alloc((size_t)Bsz * H * 4);
  float* h1a    = (float*)alloc((size_t)Bsz * H * 4);
  float* h1b    = (float*)alloc((size_t)Bsz * H * 4);
  float* x1     = (float*)alloc((size_t)Bsz * H * 4);
  float* x2     = (float*)alloc((size_t)Bsz * H * 4);
  float* catlin = (float*)alloc((size_t)Bsz * 2 * H * 4);
  float* xe     = (float*)alloc((size_t)Bsz * E * 4);              // doubles as GRU0 input
  float* logits = (float*)alloc((size_t)Bsz * V * 4);
  if (off > ws_size) return;  // workspace too small: bail (cannot allocate here)

  // ---- pack all weights to bf16 WMMA-B fragment order (L2-resident ~137MB) --
  auto packB = [&](const float* W, __bf16* dst, int K, int Nsrc, int Ntot, int nOff, int tr) {
    long total = (long)K * Nsrc;
    pack_b_kernel<<<dim3((unsigned)((total + 255) / 256)), dim3(256), 0, stream>>>(
        W, dst, K, Nsrc, Ntot, nOff, tr);
  };
  packB(gru0_wih, pk_g0_wih, E, 3 * H, 3 * H, 0, 1);
  packB(gru0_whh, pk_g0_whh, H, 3 * H, 3 * H, 0, 1);
  packB(gru1_wih, pk_g1_wih, H, 3 * H, 3 * H, 0, 1);
  packB(gru1_whh, pk_g1_whh, H, 3 * H, 3 * H, 0, 1);
  for (int k = 0; k < 2; ++k) {
    __bf16* d0 = pk_hw0_cat + (size_t)k * H * 2 * H;
    __bf16* d1 = pk_hw1_cat + (size_t)k * H * 2 * H;
    packB(hw0_wg + (size_t)k * H * H, d0, H, H, 2 * H, 0, 1);
    packB(hw0_wh + (size_t)k * H * H, d0, H, H, 2 * H, H, 1);
    packB(hw1_wg + (size_t)k * H * H, d1, H, H, 2 * H, 0, 1);
    packB(hw1_wh + (size_t)k * H * H, d1, H, H, 2 * H, H, 1);
    concat_bias_kernel<<<dim3((unsigned)((2 * H + 255) / 256)), dim3(256), 0, stream>>>(
        hw0_bg + (size_t)k * H, hw0_bh + (size_t)k * H, hwb0 + (size_t)k * 2 * H, H);
    concat_bias_kernel<<<dim3((unsigned)((2 * H + 255) / 256)), dim3(256), 0, stream>>>(
        hw1_bg + (size_t)k * H, hw1_bh + (size_t)k * H, hwb1 + (size_t)k * 2 * H, H);
  }
  packB(fc_w, pk_fc, H, V, V, 0, 1);   // fc_w is (V,H): B = fc_w^T, K=H, N=V
  packB(emb,  pk_emb, V, E, E, 0, 0);  // emb is (V,E): B = emb,    K=V, N=E

  auto packA = [&](const float* A, int K) {
    long total = (long)Bsz * K;
    pack_a_kernel<<<dim3((unsigned)((total + 255) / 256)), dim3(256), 0, stream>>>(
        A, apk, Bsz, K);
  };
  auto gemm = [&](const __bf16* Bp, const float* bias, float* C, long ldc,
                  int N, int K, int ks, int atomic) {
    gemm_bf16_kernel<<<dim3((unsigned)(N / 128), (unsigned)(Bsz / 32), (unsigned)ks),
                       dim3(256), 0, stream>>>(apk, Bp, bias, C, ldc, Bsz, N, K, atomic);
  };

  // ---- initial state ----
  hipMemcpyAsync(xe,  x0, (size_t)Bsz * E * 4, hipMemcpyDeviceToDevice, stream);
  hipMemcpyAsync(h0a, z,  (size_t)Bsz * H * 4, hipMemcpyDeviceToDevice, stream);
  fill_zero_kernel<<<dim3((unsigned)((Bsz * H + 255) / 256)), dim3(256), 0, stream>>>(
      h1a, (long)Bsz * H);

  float *h0c = h0a, *h0n = h0b, *h1c = h1a, *h1n = h1b;
  const dim3 ew_grid((unsigned)(H / 256), (unsigned)Bsz), ew_blk(256);

  for (int s = 0; s < S; ++s) {
    // ---- GRU0 ----
    packA(xe, E);
    gemm(pk_g0_wih, gru0_bih, gi, 3 * H, 3 * H, E, 1, 0);
    packA(h0c, H);
    gemm(pk_g0_whh, gru0_bhh, gh, 3 * H, 3 * H, H, 1, 0);
    gru_update_kernel<<<ew_grid, ew_blk, 0, stream>>>(gi, gh, h0c, h0n, x1, H);
    { float* t = h0c; h0c = h0n; h0n = t; }
    // ---- highway stack 0 (fused wg|wh GEMM, N = 2H) ----
    for (int k = 0; k < 2; ++k) {
      packA(x1, H);
      gemm(pk_hw0_cat + (size_t)k * H * 2 * H, hwb0 + (size_t)k * 2 * H,
           catlin, 2 * H, 2 * H, H, 1, 0);
      highway_update_kernel<<<ew_grid, ew_blk, 0, stream>>>(catlin, x1, H);
    }
    // ---- GRU1 ----
    packA(x1, H);
    gemm(pk_g1_wih, gru1_bih, gi, 3 * H, 3 * H, H, 1, 0);
    packA(h1c, H);
    gemm(pk_g1_whh, gru1_bhh, gh, 3 * H, 3 * H, H, 1, 0);
    gru_update_kernel<<<ew_grid, ew_blk, 0, stream>>>(gi, gh, h1c, h1n, x2, H);
    { float* t = h1c; h1c = h1n; h1n = t; }
    // ---- highway stack 1 ----
    for (int k = 0; k < 2; ++k) {
      packA(x2, H);
      gemm(pk_hw1_cat + (size_t)k * H * 2 * H, hwb1 + (size_t)k * 2 * H,
           catlin, 2 * H, 2 * H, H, 1, 0);
      highway_update_kernel<<<ew_grid, ew_blk, 0, stream>>>(catlin, x2, H);
    }
    // ---- FC logits (dominant GEMM: 128x32000x1024) ----
    packA(x2, H);
    gemm(pk_fc, fc_b, logits, V, V, H, 1, 0);
    // ---- gumbel softmax (in-place: logits -> probs) ----
    softmax_gumbel_kernel<<<dim3((unsigned)Bsz), dim3(256), 0, stream>>>(
        logits, gumbel + (size_t)s * Bsz * V, V);
    // ---- xe = probs @ emb, split-K over 64 slices with f32 atomics ----
    packA(logits, V);
    fill_zero_kernel<<<dim3((unsigned)((Bsz * E + 255) / 256)), dim3(256), 0, stream>>>(
        xe, (long)Bsz * E);
    gemm(pk_emb, nullptr, xe, E, E, V, 64, 1);
    // ---- write output in (B, S, E) transposed order ----
    scatter_out_kernel<<<dim3((unsigned)(E / 256), (unsigned)Bsz), dim3(256), 0, stream>>>(
        xe, out, s, S, E);
  }
}
